// RFDN_panet_61847529062480
// MI455X (gfx1250) — matrix operand order repaired and checked by reading, not compiled, not run on hardware
//
#include <hip/hip_runtime.h>
#include <stdint.h>

#define DI __device__ __forceinline__

typedef __attribute__((ext_vector_type(16))) __bf16 v16bf;
typedef __attribute__((ext_vector_type(8)))  float  v8f;
typedef __attribute__((ext_vector_type(4)))  unsigned int v4u;
typedef __attribute__((ext_vector_type(8)))  int          v8i;
typedef __attribute__((ext_vector_type(4)))  int          v4i;

// ---------------- problem constants ----------------
#define BATCH 4
#define CH    50
#define FCH   25
#define H0    48
#define W0    48
#define P0    (H0*W0)           // 2304 query positions per batch
#define LTOT  7470              // total keys across 5 scales
#define LPAD  7488              // padded to multiple of 32
#define KD    256               // padded patch dim (225 -> 256)
#define KDR   225
#define NVP   464               // padded value dim (450 -> 29*16)
#define SM_SCALE 10.0f
#define EPS_NAN  1e-4f

// per-batch strides in the fp32 workspace regions
#define REF_BSTRIDE  258300     // 50 * (1849+1444+1089+784)
#define MB_BSTRIDE   57600      // 25 * 2304
#define REFM_BSTRIDE 186750     // 25 * 7470
#define BASE_BSTRIDE 373500     // 50 * 7470

// workspace layout (element offsets)
static constexpr size_t OFF_REF  = 0;
static constexpr size_t OFF_MB   = 1033200;   // OFF_REF  + 4*REF_BSTRIDE
static constexpr size_t OFF_REFM = 1263600;   // OFF_MB   + 4*MB_BSTRIDE
static constexpr size_t OFF_BASE = 2010600;   // OFF_REFM + 4*REFM_BSTRIDE
static constexpr size_t OFF_TMP  = 3504600;   // OFF_BASE + 4*BASE_BSTRIDE
static constexpr size_t US_BYTE  = 31123456;  // 256B-aligned end of fp32 region
static constexpr size_t OFF_QU   = 0;         // 4*2304*256 ushorts
static constexpr size_t OFF_KU   = 2359296;   // 4*7488*256 ushorts
static constexpr size_t OFF_VU   = 10027008;  // 4*464*7488 ushorts

__device__ const int c_hs[5]     = {48, 43, 38, 33, 28};
__device__ const int c_area[5]   = {2304, 1849, 1444, 1089, 784};
__device__ const int c_loff[5]   = {0, 2304, 4153, 5597, 6686};
__device__ const int c_refpre[5] = {0, 0, 1849, 3293, 4382};

DI int find_scale(int site) {
    if (site < 2304) return 0;
    if (site < 4153) return 1;
    if (site < 5597) return 2;
    if (site < 6686) return 3;
    return 4;
}

DI unsigned short f2bf(float f) {  // round-to-nearest-even fp32 -> bf16
    union { float f; unsigned u; } c; c.f = f;
    unsigned r = c.u + 0x7FFFu + ((c.u >> 16) & 1u);
    return (unsigned short)(r >> 16);
}

DI float keys_cubic(float t) {     // Keys cubic, a = -0.5
    t = fabsf(t);
    float t2 = t * t, t3 = t2 * t;
    if (t <= 1.f) return 1.5f * t3 - 2.5f * t2 + 1.f;
    if (t <  2.f) return -0.5f * t3 + 2.5f * t2 - 4.f * t + 2.f;
    return 0.f;
}

// ---- Tensor Data Mover: 2D tile (rows x 8B-units) global -> LDS (ISA 08, D# groups) ----
// amdgpu-toolchain (clang-23) 6-arg form: (g0, g1, g2, g3, g4, cpol)
DI void tdm_load_2d(unsigned lds_byte, unsigned long long ga,
                    unsigned tile_d0_u8, unsigned tile_d1, unsigned stride0_u8) {
    v4u g0;
    g0[0] = 1u;                                   // count=1, user descriptor
    g0[1] = lds_byte;                             // lds_addr (bytes)
    g0[2] = (unsigned)ga;                         // global_addr[31:0]
    g0[3] = ((unsigned)(ga >> 32) & 0x01FFFFFFu)  // global_addr[56:32]
            | 0x80000000u;                        // type = 2 ("image")
    v8i g1;
    g1[0] = (int)(3u << 16);                      // data_size = 3 (8 bytes); no mcast/pad
    g1[1] = (int)0xFFFF0000u;                     // tensor_dim0 = 0x7FFFFFFF (lo16 << 16)
    g1[2] = (int)0xFFFF7FFFu;                     // tensor_dim0 hi16 | tensor_dim1 lo16<<16
    g1[3] = (int)(0x7FFFu | (tile_d0_u8 << 16));  // tensor_dim1 hi16 | tile_dim0
    g1[4] = (int)(tile_d1 & 0xFFFFu);             // tile_dim1 | tile_dim2=0
    g1[5] = (int)stride0_u8;                      // tensor_dim0_stride lo32 (8B units)
    g1[6] = 0;                                    // stride0 hi | stride1 lo
    g1[7] = 0;                                    // stride1 hi
    v4i z4 = {0, 0, 0, 0};
    v8i z8 = {0, 0, 0, 0, 0, 0, 0, 0};
    __builtin_amdgcn_tensor_load_to_lds(g0, g1, z4, z4, z8, 0);
}

// ---------------- bicubic (antialiased) resize, one scale per launch ----------------
__global__ void k_resize(const float* __restrict__ x, float* __restrict__ refseg,
                         int Hs, int Ws, float sc, int N) {
    int tid = blockIdx.x * blockDim.x + threadIdx.x;
    if (tid >= N) return;
    int area = Hs * Ws;
    int b  = tid / (CH * area);
    int r  = tid % (CH * area);
    int c  = r / area;
    int li = r % area;
    int oy = li / Ws, ox = li % Ws;
    const float* xin = x + ((size_t)b * CH + c) * (size_t)P0;
    float uy = (oy + 0.5f) / sc - 0.5f;
    float ux = (ox + 0.5f) / sc - 0.5f;
    float sup = 2.0f / sc;
    int ylo = (int)ceilf(uy - sup), yhi = (int)floorf(uy + sup);
    int xlo = (int)ceilf(ux - sup), xhi = (int)floorf(ux + sup);
    float acc = 0.f, wsum = 0.f;
    for (int ky = ylo; ky <= yhi; ++ky) {
        if (ky < 0 || ky >= H0) continue;
        float wy = keys_cubic((ky - uy) * sc);
        for (int kx = xlo; kx <= xhi; ++kx) {
            if (kx < 0 || kx >= W0) continue;
            float wx = keys_cubic((kx - ux) * sc);
            acc  += wy * wx * xin[ky * W0 + kx];
            wsum += wy * wx;
        }
    }
    refseg[(size_t)b * REF_BSTRIDE + (size_t)c * area + li] = acc / wsum;
}

// ---------------- 1x1 convs + PReLU at every pyramid site ----------------
__global__ void k_proj(const float* __restrict__ x, const float* __restrict__ ref,
                       const float* wbase, const float* bbase, const float* abase,
                       const float* wmatch, const float* bmatch, const float* amatch,
                       const float* wasm, const float* basm, const float* aasm,
                       float* __restrict__ mb, float* __restrict__ refm,
                       float* __restrict__ basef, int N) {
    int tid = blockIdx.x * blockDim.x + threadIdx.x;
    if (tid >= N) return;
    int b = tid / LTOT, site = tid % LTOT;
    int s = find_scale(site);
    int li = site - c_loff[s];
    int area = c_area[s];
    const float* in = (s == 0) ? (x + (size_t)b * CH * P0)
                               : (ref + (size_t)b * REF_BSTRIDE + (size_t)CH * c_refpre[s]);
    float v[CH];
    for (int i = 0; i < CH; ++i) v[i] = in[(size_t)i * area + li];

    float am = amatch[0];
    float* rm = refm + (size_t)b * REFM_BSTRIDE + (size_t)FCH * c_loff[s];
    for (int o = 0; o < FCH; ++o) {
        float acc = bmatch[o];
        for (int i = 0; i < CH; ++i) acc += wmatch[o * CH + i] * v[i];
        rm[(size_t)o * area + li] = (acc >= 0.f) ? acc : am * acc;
    }
    float aa = aasm[0];
    float* bs = basef + (size_t)b * BASE_BSTRIDE + (size_t)CH * c_loff[s];
    for (int o = 0; o < CH; ++o) {
        float acc = basm[o];
        for (int i = 0; i < CH; ++i) acc += wasm[o * CH + i] * v[i];
        bs[(size_t)o * area + li] = (acc >= 0.f) ? acc : aa * acc;
    }
    if (s == 0) {
        float ab = abase[0];
        float* m = mb + (size_t)b * MB_BSTRIDE;
        for (int o = 0; o < FCH; ++o) {
            float acc = bbase[o];
            for (int i = 0; i < CH; ++i) acc += wbase[o * CH + i] * v[i];
            m[(size_t)o * P0 + li] = (acc >= 0.f) ? acc : ab * acc;
        }
    }
}

// ---------------- Q matrix: patches of match_base -> bf16 [B][P0][KD] ----------------
__global__ void k_buildQ(const float* __restrict__ mb, unsigned short* __restrict__ Qm, int N) {
    int tid = blockIdx.x * blockDim.x + threadIdx.x;
    if (tid >= N) return;
    int b = tid / P0, q = tid % P0;
    int y = q / W0, x = q % W0;
    const float* m = mb + (size_t)b * MB_BSTRIDE;
    unsigned short* qr = Qm + ((size_t)b * P0 + q) * KD;
    for (int c = 0; c < FCH; ++c)
        for (int dy = 0; dy < 3; ++dy) {
            int yy = y + dy - 1;
            for (int dx = 0; dx < 3; ++dx) {
                int xx = x + dx - 1;
                float v = 0.f;
                if (yy >= 0 && yy < H0 && xx >= 0 && xx < W0) v = m[(size_t)c * P0 + yy * W0 + xx];
                qr[c * 9 + dy * 3 + dx] = f2bf(v);
            }
        }
    for (int k = KDR; k < KD; ++k) qr[k] = 0;
}

// ---------------- K matrix: normalized refm patches * SM_SCALE -> bf16 [B][LPAD][KD] ----------------
__global__ void k_buildK(const float* __restrict__ refm, unsigned short* __restrict__ Km, int N) {
    int tid = blockIdx.x * blockDim.x + threadIdx.x;
    if (tid >= N) return;
    int b = tid / LPAD, l = tid % LPAD;
    unsigned short* kr = Km + ((size_t)b * LPAD + l) * KD;
    if (l >= LTOT) { for (int k = 0; k < KD; ++k) kr[k] = 0; return; }
    int s = find_scale(l);
    int li = l - c_loff[s];
    int Hs = c_hs[s], Ws = c_hs[s], area = c_area[s];
    int y = li / Ws, x = li % Ws;
    const float* rm = refm + (size_t)b * REFM_BSTRIDE + (size_t)FCH * c_loff[s];
    float ss = 0.f;
    for (int c = 0; c < FCH; ++c)
        for (int dy = 0; dy < 3; ++dy) {
            int yy = y + dy - 1;
            for (int dx = 0; dx < 3; ++dx) {
                int xx = x + dx - 1;
                float v = 0.f;
                if (yy >= 0 && yy < Hs && xx >= 0 && xx < Ws) v = rm[(size_t)c * area + yy * Ws + xx];
                ss += v * v;
            }
        }
    float scl = SM_SCALE / fmaxf(sqrtf(ss), EPS_NAN);
    for (int c = 0; c < FCH; ++c)
        for (int dy = 0; dy < 3; ++dy) {
            int yy = y + dy - 1;
            for (int dx = 0; dx < 3; ++dx) {
                int xx = x + dx - 1;
                float v = 0.f;
                if (yy >= 0 && yy < Hs && xx >= 0 && xx < Ws) v = rm[(size_t)c * area + yy * Ws + xx];
                kr[c * 9 + dy * 3 + dx] = f2bf(v * scl);
            }
        }
    for (int k = KDR; k < KD; ++k) kr[k] = 0;
}

// ---------------- V matrix (transposed): base patches -> bf16 [B][NVP][LPAD] ----------------
__global__ void k_buildV(const float* __restrict__ basef, unsigned short* __restrict__ Vt, int N) {
    int tid = blockIdx.x * blockDim.x + threadIdx.x;
    if (tid >= N) return;
    int b = tid / LPAD, l = tid % LPAD;
    if (l >= LTOT) {
        for (int j = 0; j < NVP; ++j) Vt[((size_t)b * NVP + j) * LPAD + l] = 0;
        return;
    }
    int s = find_scale(l);
    int li = l - c_loff[s];
    int Hs = c_hs[s], Ws = c_hs[s], area = c_area[s];
    int y = li / Ws, x = li % Ws;
    const float* bs = basef + (size_t)b * BASE_BSTRIDE + (size_t)CH * c_loff[s];
    for (int c = 0; c < CH; ++c)
        for (int ky = 0; ky < 3; ++ky) {
            int yy = y + ky - 1;
            for (int kx = 0; kx < 3; ++kx) {
                int xx = x + kx - 1;
                float v = 0.f;
                if (yy >= 0 && yy < Hs && xx >= 0 && xx < Ws) v = bs[(size_t)c * area + yy * Ws + xx];
                Vt[((size_t)b * NVP + c * 9 + ky * 3 + kx) * LPAD + l] = f2bf(v);
            }
        }
    for (int j = 450; j < NVP; ++j) Vt[((size_t)b * NVP + j) * LPAD + l] = 0;
}

// ---------------- flash attention with TDM double-buffered K/V staging ----------------
__global__ __launch_bounds__(128) void k_flash(const unsigned short* __restrict__ Qm,
                                               const unsigned short* __restrict__ Km,
                                               const unsigned short* __restrict__ Vt,
                                               float* __restrict__ tmp) {
    __shared__ unsigned short shK[2 * 32 * KD];     // 2 x 16KB  : K tile, 32 keys x 256 kdim
    __shared__ unsigned short shV[2 * NVP * 32];    // 2 x 29KB  : V tile, 464 cols x 32 keys
    __shared__ unsigned short shP[4 * 512];         // per-wave P transpose staging

    const int lane = threadIdx.x & 31;
    const int w    = threadIdx.x >> 5;
    const int tile = blockIdx.x * 2 + (w >> 1);     // 2 q-tiles per block, 2 waves per tile
    const int half = w & 1;                         // V-column slice of this wave
    const int b    = tile / (P0 / 16);
    const int q0   = (tile % (P0 / 16)) * 16;
    const int n    = lane & 15;
    const int hiL  = lane >> 4;
    const int kb8  = hiL * 8;
    const int kh16 = hiL * 16;
    const int tn   = half ? 14 : 15;                // 29 = 15 + 14 V tiles

    union FragBF { v16bf v; uint4 u4[2]; };

    // Q A-fragments (resident for the whole kernel)
    FragBF qf[8];
    const unsigned short* qrow = Qm + ((size_t)b * P0 + q0 + n) * KD;
#pragma unroll
    for (int c = 0; c < 8; ++c) {
        qf[c].u4[0] = *(const uint4*)(qrow + c * 32 + kb8);
        qf[c].u4[1] = *(const uint4*)(qrow + c * 32 + 16 + kb8);
    }

    v8f o[15];
#pragma unroll
    for (int t = 0; t < 15; ++t)
#pragma unroll
        for (int i = 0; i < 8; ++i) o[t][i] = 0.f;
    float mrow[8], lrow[8];
#pragma unroll
    for (int i = 0; i < 8; ++i) { mrow[i] = -1e30f; lrow[i] = 0.f; }

    unsigned short* ps = shP + w * 512;
    const unsigned ldsK = (unsigned)(uintptr_t)(&shK[0]);
    const unsigned ldsV = (unsigned)(uintptr_t)(&shV[0]);
    const int NIT = LPAD / 32;                      // 234

    // prologue: DMA tile 0 into buffer 0 (wave 0 drives the TDM)
    if (w == 0) {
        unsigned long long gaK = (unsigned long long)(uintptr_t)(Km + ((size_t)b * LPAD) * KD);
        unsigned long long gaV = (unsigned long long)(uintptr_t)(Vt + ((size_t)b * NVP) * LPAD);
        tdm_load_2d(ldsK, gaK, 2048u, 1u, 2048u);           // 16KB contiguous
        tdm_load_2d(ldsV, gaV, 8u, (unsigned)NVP, 1872u);   // 464 rows x 64B, stride LPAD*2
    }

    for (int it = 0; it < NIT; ++it) {
        const int buf = it & 1;
        const int l0  = it * 32;
        // prefetch tile it+1 into the other buffer (overlaps this whole iteration)
        if (w == 0) {
            if (it + 1 < NIT) {
                const int l0n = l0 + 32;
                unsigned long long gaK = (unsigned long long)(uintptr_t)
                    (Km + ((size_t)b * LPAD + l0n) * KD);
                unsigned long long gaV = (unsigned long long)(uintptr_t)
                    (Vt + ((size_t)b * NVP) * LPAD + l0n);
                tdm_load_2d(ldsK + (buf ^ 1) * (32 * KD * 2), gaK, 2048u, 1u, 2048u);
                tdm_load_2d(ldsV + (buf ^ 1) * (NVP * 32 * 2), gaV, 8u, (unsigned)NVP, 1872u);
                __builtin_amdgcn_s_wait_tensorcnt(2);       // tile `it` complete
            } else {
                __builtin_amdgcn_s_wait_tensorcnt(0);
            }
        }
        __syncthreads();                                    // tile `it` visible to all waves

        // ---- S = Q * K^T for 32 keys (two 16-key column tiles), K from LDS ----
        v8f s0, s1;
#pragma unroll
        for (int i = 0; i < 8; ++i) { s0[i] = 0.f; s1[i] = 0.f; }
        const unsigned short* kr0 = shK + buf * (32 * KD) + n * KD + kb8;
        const unsigned short* kr1 = kr0 + 16 * KD;
#pragma unroll
        for (int c = 0; c < 8; ++c) {
            FragBF bk0, bk1;
            bk0.u4[0] = *(const uint4*)(kr0 + c * 32);
            bk0.u4[1] = *(const uint4*)(kr0 + c * 32 + 16);
            bk1.u4[0] = *(const uint4*)(kr1 + c * 32);
            bk1.u4[1] = *(const uint4*)(kr1 + c * 32 + 16);
            s0 = __builtin_amdgcn_wmma_f32_16x16x32_bf16(false, qf[c].v, false, bk0.v, (short)0, s0, false, false);
            s1 = __builtin_amdgcn_wmma_f32_16x16x32_bf16(false, qf[c].v, false, bk1.v, (short)0, s1, false, false);
        }
        // ---- online softmax ----
        const bool val0 = (l0 + n) < LTOT;
        const bool val1 = (l0 + 16 + n) < LTOT;
#pragma unroll
        for (int i = 0; i < 8; ++i) {
            float a0 = val0 ? s0[i] : -1e30f;
            float a1 = val1 ? s1[i] : -1e30f;
            float tm = fmaxf(a0, a1);
            tm = fmaxf(tm, __shfl_xor(tm, 1, 32));
            tm = fmaxf(tm, __shfl_xor(tm, 2, 32));
            tm = fmaxf(tm, __shfl_xor(tm, 4, 32));
            tm = fmaxf(tm, __shfl_xor(tm, 8, 32));
            float nm = fmaxf(mrow[i], tm);
            float cr = __expf(mrow[i] - nm);
            float e0 = __expf(a0 - nm);
            float e1 = __expf(a1 - nm);
            float rs = e0 + e1;
            rs += __shfl_xor(rs, 1, 32);
            rs += __shfl_xor(rs, 2, 32);
            rs += __shfl_xor(rs, 4, 32);
            rs += __shfl_xor(rs, 8, 32);
            lrow[i] = lrow[i] * cr + rs;
            mrow[i] = nm;
#pragma unroll
            for (int t = 0; t < 15; ++t) if (t < tn) o[t][i] *= cr;
            int row = kb8 + i;
            ps[row * 32 + n]      = f2bf(e0);
            ps[row * 32 + 16 + n] = f2bf(e1);
        }
        __syncthreads();                                    // P staged
        // ---- reload P as an A-fragment (in-wave transpose via LDS) ----
        FragBF pf;
        pf.u4[0] = *(const uint4*)(ps + n * 32 + kb8);
        pf.u4[1] = *(const uint4*)(ps + n * 32 + 16 + kb8);
        // ---- O += P * V over this wave's V column slice, V from LDS ----
#pragma unroll
        for (int t = 0; t < 15; ++t) if (t < tn) {
            int j = (half * 15 + t) * 16 + n;
            const unsigned short* vr = shV + buf * (NVP * 32) + j * 32 + kb8;
            FragBF bv;
            bv.u4[0] = *(const uint4*)(vr);
            bv.u4[1] = *(const uint4*)(vr + 16);
            o[t] = __builtin_amdgcn_wmma_f32_16x16x32_bf16(false, pf.v, false, bv.v, (short)0, o[t], false, false);
        }
        __syncthreads();                                    // buffer `buf` fully consumed
    }
    // ---- normalize and store ----
#pragma unroll
    for (int i = 0; i < 8; ++i) lrow[i] = 1.0f / fmaxf(lrow[i], 1e-20f);
#pragma unroll
    for (int t = 0; t < 15; ++t) if (t < tn) {
#pragma unroll
        for (int i = 0; i < 8; ++i) {
            int row = q0 + kb8 + i;
            int col = (half * 15 + t) * 16 + n;
            tmp[((size_t)b * P0 + row) * NVP + col] = o[t][i] * lrow[i];
        }
    }
}

// ---------------- fold (conv_transpose epilogue) + residual ----------------
__global__ void k_fold(const float* __restrict__ x, const float* __restrict__ tmp,
                       float* __restrict__ out, int N) {
    int tid = blockIdx.x * blockDim.x + threadIdx.x;
    if (tid >= N) return;
    int b = tid / (CH * P0);
    int r = tid % (CH * P0);
    int c = r / P0;
    int p = r % P0;
    int y = p / W0, xx = p % W0;
    float acc = 0.f;
    for (int ky = 0; ky < 3; ++ky) {
        int yy = y + ky - 1;
        if (yy < 0 || yy >= H0) continue;
        for (int kx = 0; kx < 3; ++kx) {
            int xc = xx + kx - 1;
            if (xc < 0 || xc >= W0) continue;
            int q = yy * W0 + xc;
            int j = c * 9 + (2 - ky) * 3 + (2 - kx);
            acc += tmp[((size_t)b * P0 + q) * NVP + j];
        }
    }
    out[tid] = x[tid] + 0.25f * acc;
}

// ---------------- host launch ----------------
extern "C" void kernel_launch(void* const* d_in, const int* in_sizes, int n_in,
                              void* d_out, int out_size, void* d_ws, size_t ws_size,
                              hipStream_t stream) {
    const float* x       = (const float*)d_in[0];
    const float* w_base  = (const float*)d_in[1];
    const float* b_base  = (const float*)d_in[2];
    const float* a_base  = (const float*)d_in[3];
    const float* w_match = (const float*)d_in[4];
    const float* b_match = (const float*)d_in[5];
    const float* a_match = (const float*)d_in[6];
    const float* w_asm   = (const float*)d_in[7];
    const float* b_asm   = (const float*)d_in[8];
    const float* a_asm   = (const float*)d_in[9];

    float* wsf = (float*)d_ws;
    unsigned short* wsu = (unsigned short*)((char*)d_ws + US_BYTE);
    float* ref   = wsf + OFF_REF;
    float* mb    = wsf + OFF_MB;
    float* refm  = wsf + OFF_REFM;
    float* basef = wsf + OFF_BASE;
    float* tmp   = wsf + OFF_TMP;
    unsigned short* Qm = wsu + OFF_QU;
    unsigned short* Km = wsu + OFF_KU;
    unsigned short* Vt = wsu + OFF_VU;

    const int hs[5]     = {48, 43, 38, 33, 28};
    const int area[5]   = {2304, 1849, 1444, 1089, 784};
    const int refpre[5] = {0, 0, 1849, 3293, 4382};

    for (int s = 1; s < 5; ++s) {
        int N = BATCH * CH * area[s];
        k_resize<<<(N + 255) / 256, 256, 0, stream>>>(
            x, ref + (size_t)CH * refpre[s], hs[s], hs[s], (float)hs[s] / 48.0f, N);
    }
    {
        int N = BATCH * LTOT;
        k_proj<<<(N + 255) / 256, 256, 0, stream>>>(
            x, ref, w_base, b_base, a_base, w_match, b_match, a_match,
            w_asm, b_asm, a_asm, mb, refm, basef, N);
    }
    {
        int N = BATCH * P0;
        k_buildQ<<<(N + 255) / 256, 256, 0, stream>>>(mb, Qm, N);
    }
    {
        int N = BATCH * LPAD;
        k_buildK<<<(N + 255) / 256, 256, 0, stream>>>(refm, Km, N);
        k_buildV<<<(N + 255) / 256, 256, 0, stream>>>(basef, Vt, N);
    }
    {
        int blocks = (BATCH * (P0 / 16)) / 2;   // 288 blocks, 2 q-tiles each
        k_flash<<<blocks, 128, 0, stream>>>(Qm, Km, Vt, tmp);
    }
    {
        int N = BATCH * CH * P0;
        k_fold<<<(N + 255) / 256, 256, 0, stream>>>(x, tmp, (float*)d_out, N);
    }
}
